// RoutingModule_13331578486932
// MI455X (gfx1250) — compile-verified, matrix operand order checked by reading, not run on hardware
//
#include <hip/hip_runtime.h>
#include <cmath>

// ---------------------------------------------------------------------------
// RoutingModule on MI455X (gfx1250, wave32, WMMA + Tensor Data Mover).
//
// Two 16384x2048x2048 GEMMs (proj = x@wq^T, k = x@wk^T) in fp16 WMMA with
// f32 accumulation; only per-row reductions (||proj||^2, ||k||^2,
// proj[i].k[i+1]) leave the GEMM via f32 atomics. Data path per CDNA5:
//   global --TDM(tensor_load_to_lds, padded)--> LDS --ds_load_b128--> WMMA
// double-buffered on TENSORcnt + one workgroup barrier per K-stage.
// fp16 x + weights (~84 MB) stay resident in the 192 MB L2.
// ---------------------------------------------------------------------------

typedef _Float16 v8h  __attribute__((ext_vector_type(8)));
typedef _Float16 v16h __attribute__((ext_vector_type(16)));
typedef float    v8f  __attribute__((ext_vector_type(8)));
typedef unsigned int v4u __attribute__((ext_vector_type(4)));
typedef int      v4i  __attribute__((ext_vector_type(4)));
typedef int      v8i  __attribute__((ext_vector_type(8)));

#define BM 128      // block rows
#define BN 128      // block cols
#define KS 64       // K halves staged per TDM buffer (2 WMMA substeps)
#define ASTRIDE 72  // LDS row stride in halves: 64 data + 8 pad (144 B)
#define TILE_H (BM * ASTRIDE)  // halves per staged tile (A or B)

// Load one 16x32 fp16 A/B fragment per the CDNA5 ISA VGPR layout:
// lane L holds row (L&15); lanes 0-15 carry K=[0..7]+[16..23],
// lanes 16-31 carry K=[8..15]+[24..31]. Two 16-byte ds loads.
__device__ __forceinline__ v16h load_frag16(const _Float16* p) {
  v8h lo = *(const v8h*)(p);
  v8h hi = *(const v8h*)(p + 16);
  return __builtin_shufflevector(lo, hi, 0, 1, 2, 3, 4, 5, 6, 7,
                                 8, 9, 10, 11, 12, 13, 14, 15);
}

__device__ __forceinline__ v8f wmma_f16(v16h a, v16h b, v8f c) {
  return __builtin_amdgcn_wmma_f32_16x16x32_f16(
      false, a, false, b, (short)0, c, false, false);
}

__device__ __forceinline__ float hgroup_sum16(float s) {
  s += __shfl_xor(s, 1);
  s += __shfl_xor(s, 2);
  s += __shfl_xor(s, 4);
  s += __shfl_xor(s, 8);
  return s;
}

// Issue one TDM 2D tile load: BM rows x KS halves from row-major fp16 source
// (row stride = D halves) into LDS at lds_byte_off, padding +16B per 128B row
// so LDS rows land at ASTRIDE halves. data_size = 8B units.
__device__ __forceinline__ void tdm_stage(unsigned lds_byte_off,
                                          const _Float16* gbase,
                                          int stride_units /* D/4 */) {
  unsigned long long ga = (unsigned long long)gbase;
  v4u g0;
  g0.x = 1u;                                   // count=1, user descriptor
  g0.y = lds_byte_off;                         // lds_addr
  g0.z = (unsigned)(ga & 0xFFFFFFFFu);         // global_addr[31:0]
  g0.w = (unsigned)((ga >> 32) & 0x1FFFFFFu)   // global_addr[56:32]
         | (2u << 30);                         // type=2 ("image")
  v8i g1;
  g1[0] = (3 << 16)      // data_size = 8B
        | (1 << 20)      // pad_enable
        | (4 << 22)      // pad_interval: 32 DWORDs (128B) before pad
        | (3 << 25);     // pad_amount: 4 DWORDs (16B)
  g1[1] = (int)(((unsigned)stride_units & 0xFFFFu) << 16);        // tensor_dim0 lo16
  g1[2] = (int)((((unsigned)stride_units >> 16) & 0xFFFFu)        // tensor_dim0 hi16
                | ((unsigned)BM << 16));                          // tensor_dim1 lo16
  g1[3] = (int)((16u << 16));                  // tensor_dim1 hi16=0 | tile_dim0=16 units (128B)
  g1[4] = BM;                                  // tile_dim1=BM rows, tile_dim2=0
  g1[5] = stride_units;                        // tensor_dim0_stride lo32 (D/4 units)
  g1[6] = 0;                                   // stride hi | tensor_dim1_stride lo (unused, 2D)
  g1[7] = 0;
  v4i z4 = {0, 0, 0, 0};
#if __clang_major__ >= 23
  v8i z8 = {0, 0, 0, 0, 0, 0, 0, 0};
  __builtin_amdgcn_tensor_load_to_lds(g0, g1, z4, z4, z8, 0);
#else
  __builtin_amdgcn_tensor_load_to_lds(g0, g1, z4, z4, 0);
#endif
}

// MODE 0: proj pass -> store proj fp16, accumulate qn2[row]
// MODE 1: k    pass -> accumulate kn2[row], dp[row-1] += proj[row-1].k[row]
template <int MODE>
__global__ __launch_bounds__(256)
void gemm_rowred(const _Float16* __restrict__ xh,
                 const _Float16* __restrict__ wh,
                 _Float16* __restrict__ projh,
                 float* __restrict__ rown2,
                 float* __restrict__ dpv,
                 int T, int D) {
  __shared__ _Float16 smem[4 * TILE_H];  // {A0, B0, A1, B1}, 72 KB

  const int lane = threadIdx.x & 31;
  const int wave = threadIdx.x >> 5;
  const int wm = wave & 3;   // 4 row-waves  -> 32 rows each
  const int wn = wave >> 2;  // 2 col-waves  -> 64 cols each
  const int i0 = blockIdx.y * BM;
  const int j0 = blockIdx.x * BN;

  const int r    = lane & 15;
  const int ksel = (lane >> 4) << 3;  // 0 or 8
  const int su   = D >> 2;            // row stride in 8B units

  typedef __attribute__((address_space(3))) void as3_void;
  const unsigned lds_base = (unsigned)(unsigned long long)(as3_void*)smem;

  const _Float16* gA = xh + (size_t)i0 * D;  // A tile rows (x)
  const _Float16* gB = wh + (size_t)j0 * D;  // B tile rows (w): B col = w row

  const int NK = D / KS;  // staged K chunks

  if (wave == 0) {  // prologue: stage buffer 0
    tdm_stage(lds_base + 0 * TILE_H * 2, gA, su);
    tdm_stage(lds_base + 1 * TILE_H * 2, gB, su);
  }

  v8f C[2][4] = {};

  for (int kt = 0; kt < NK; ++kt) {
    if (wave == 0) __builtin_amdgcn_s_wait_tensorcnt(0);
    __syncthreads();  // current buffer visible to all; prior reads retired

    const int buf = kt & 1;
    if (wave == 0 && kt + 1 < NK) {  // stage next buffer while computing
      tdm_stage(lds_base + (2 * (buf ^ 1) + 0) * TILE_H * 2, gA + (kt + 1) * KS, su);
      tdm_stage(lds_base + (2 * (buf ^ 1) + 1) * TILE_H * 2, gB + (kt + 1) * KS, su);
    }

    const _Float16* la = &smem[(2 * buf + 0) * TILE_H + (wm * 32 + r) * ASTRIDE + ksel];
    const _Float16* lb = &smem[(2 * buf + 1) * TILE_H + (wn * 64 + r) * ASTRIDE + ksel];

#pragma unroll
    for (int kc = 0; kc < KS; kc += 32) {
      v16h a0 = load_frag16(la + kc);
      v16h a1 = load_frag16(la + 16 * ASTRIDE + kc);
      v16h b0 = load_frag16(lb + kc);
      v16h b1 = load_frag16(lb + 16 * ASTRIDE + kc);
      v16h b2 = load_frag16(lb + 32 * ASTRIDE + kc);
      v16h b3 = load_frag16(lb + 48 * ASTRIDE + kc);
      C[0][0] = wmma_f16(a0, b0, C[0][0]);
      C[0][1] = wmma_f16(a0, b1, C[0][1]);
      C[0][2] = wmma_f16(a0, b2, C[0][2]);
      C[0][3] = wmma_f16(a0, b3, C[0][3]);
      C[1][0] = wmma_f16(a1, b0, C[1][0]);
      C[1][1] = wmma_f16(a1, b1, C[1][1]);
      C[1][2] = wmma_f16(a1, b2, C[1][2]);
      C[1][3] = wmma_f16(a1, b3, C[1][3]);
    }
  }

  // C layout: lane holds column N=(lane&15); VGPR v holds row M=v+8*(lane>=16).
  const int rowadd = (lane >> 4) << 3;

#pragma unroll
  for (int mf = 0; mf < 2; ++mf) {
#pragma unroll
    for (int v = 0; v < 8; ++v) {
      const int row = i0 + wm * 32 + mf * 16 + v + rowadd;
      float s = 0.f, d = 0.f;
#pragma unroll
      for (int nf = 0; nf < 4; ++nf) {
        const float val = C[mf][nf][v];
        const int col = j0 + wn * 64 + nf * 16 + (lane & 15);
        if (MODE == 0) projh[(size_t)row * D + col] = (_Float16)val;
        s += val * val;
        if (MODE == 1) {
          const int prow = (row > 0) ? row - 1 : 0;  // clamped; guarded below
          d += (float)projh[(size_t)prow * D + col] * val;
        }
      }
      s = hgroup_sum16(s);
      if ((lane & 15) == 0) unsafeAtomicAdd(&rown2[row], s);
      if (MODE == 1) {
        d = hgroup_sum16(d);
        if ((lane & 15) == 0 && row > 0) unsafeAtomicAdd(&dpv[row - 1], d);
      }
    }
  }
}

__global__ void f32_to_f16_kernel(const float* __restrict__ src,
                                  _Float16* __restrict__ dst, size_t n) {
  size_t i = (size_t)blockIdx.x * blockDim.x + threadIdx.x;
  const size_t stride = (size_t)gridDim.x * blockDim.x;
  for (; i < n; i += stride) dst[i] = (_Float16)src[i];
}

__global__ void finalize_kernel(const float* __restrict__ qn2,
                                const float* __restrict__ kn2,
                                const float* __restrict__ dpv,
                                const int* __restrict__ offs, int B, int T,
                                float* __restrict__ out) {
  const int t = blockIdx.x * blockDim.x + threadIdx.x;
  if (t >= T) return;

  bool is_start = false;
  for (int s = 0; s < B; ++s) is_start |= (offs[s] == t);

  const float eps = 1e-8f;
  const float k2  = fmaxf(kn2[t], 0.f);
  const float knc = fmaxf(sqrtf(k2), eps);

  float p;
  if (is_start) {
    p = 0.5f + 0.5f * (k2 / (knc * knc));  // q = -k exactly
  } else if (t == 0) {
    p = 0.5f;                              // q = 0 row
  } else {
    const float qnc = fmaxf(sqrtf(fmaxf(qn2[t - 1], 0.f)), eps);
    p = 0.5f - 0.5f * (dpv[t - 1] / (qnc * knc));
  }
  p = fminf(fmaxf(p, 0.f), 1.f);
  const float bsel = (p >= 0.5f) ? 1.f : 0.f;

  out[t]         = p;
  out[T + t]     = bsel;
  out[2 * T + t] = bsel * p;
}

extern "C" void kernel_launch(void* const* d_in, const int* in_sizes, int n_in,
                              void* d_out, int out_size, void* d_ws,
                              size_t ws_size, hipStream_t stream) {
  const float* x    = (const float*)d_in[0];
  const float* wq   = (const float*)d_in[1];
  const float* wk   = (const float*)d_in[2];
  const int*   offs = (const int*)d_in[3];

  const int D = (int)llround(sqrt((double)in_sizes[1]));  // 2048
  const int T = in_sizes[0] / D;                          // 16384
  const int B = in_sizes[3] - 1;                          // 8

  char* ws = (char*)d_ws;
  float* qn2 = (float*)ws;  // T
  float* kn2 = qn2 + T;     // T
  float* dpv = kn2 + T;     // T
  size_t off = (((size_t)3 * T * sizeof(float)) + 255) & ~(size_t)255;
  _Float16* xh    = (_Float16*)(ws + off);
  _Float16* wqh   = xh + (size_t)T * D;
  _Float16* wkh   = wqh + (size_t)D * D;
  _Float16* projh = wkh + (size_t)D * D;
  (void)ws_size; (void)n_in; (void)out_size;

  hipMemsetAsync(qn2, 0, (size_t)3 * T * sizeof(float), stream);

  f32_to_f16_kernel<<<4096, 256, 0, stream>>>(x, xh, (size_t)T * D);
  f32_to_f16_kernel<<<2048, 256, 0, stream>>>(wq, wqh, (size_t)D * D);
  f32_to_f16_kernel<<<2048, 256, 0, stream>>>(wk, wkh, (size_t)D * D);

  dim3 grid(D / BN, T / BM);
  gemm_rowred<0><<<grid, 256, 0, stream>>>(xh, wqh, projh, qn2, nullptr, T, D);
  gemm_rowred<1><<<grid, 256, 0, stream>>>(xh, wkh, projh, kn2, dpv, T, D);

  finalize_kernel<<<(T + 255) / 256, 256, 0, stream>>>(qn2, kn2, dpv, offs, B,
                                                       T, (float*)d_out);
}
